// BidirectionalCrossAttention_38173669327505
// MI455X (gfx1250) — matrix operand order repaired and checked by reading, not compiled, not run on hardware
//
#include <hip/hip_runtime.h>

#define HEADS 8
#define DHEAD 64
#define SEQ   2048
#define BATCH 2
#define DIMF  512
#define ROWS  (BATCH*SEQ)     // 4096
#define SCL   0.125f          // 64^-0.5
#define EPSV  1e-5f

typedef __bf16 v16bf __attribute__((ext_vector_type(16)));
typedef float  v8f   __attribute__((ext_vector_type(8)));

union Frag { v16bf v; uint4 q[2]; };

__device__ __forceinline__ unsigned short f2bf(float f) {
    union { float f; unsigned u; } x; x.f = f;
    unsigned r = x.u + 0x7FFFu + ((x.u >> 16) & 1u);   // round-to-nearest-even
    return (unsigned short)(r >> 16);
}

__device__ __forceinline__ v8f wmma_bf16(const Frag& a, const Frag& b, v8f c) {
    return __builtin_amdgcn_wmma_f32_16x16x32_bf16(
        /*neg_a=*/false, a.v, /*neg_b=*/false, b.v,
        /*c_mod=*/(short)0, c, /*reuse_a=*/false, /*reuse_b=*/false);
}

// Async copy 16B/lane from global to LDS (GLOBAL_LOAD_ASYNC_TO_LDS_B128, ASYNCcnt).
__device__ __forceinline__ void async_b128(const void* gaddr, void* lds_generic) {
    unsigned lds = (unsigned)(size_t)lds_generic;              // low 32b = LDS offset
    unsigned long long ga = (unsigned long long)(size_t)gaddr;
    asm volatile("global_load_async_to_lds_b128 %0, %1, off"
                 :: "v"(lds), "v"(ga) : "memory");
}
__device__ __forceinline__ void async_wait0() {
    asm volatile("s_wait_asynccnt 0x0" ::: "memory");
}

// ---------------------------------------------------------------- f32 -> bf16 (transposed)
// dst[n*512 + k] = bf16(src[k*512 + n])  (weights stored column-major for GEMM B tiles)
__global__ void cvt_w_transpose(const float* __restrict__ src,
                                unsigned short* __restrict__ dst) {
    int i = blockIdx.x * blockDim.x + threadIdx.x;   // i over 512*512
    int k = i >> 9, n = i & (DIMF - 1);
    dst[(size_t)n * DIMF + k] = f2bf(src[i]);
}

// ---------------------------------------------------------------- LayerNorm -> bf16
__global__ __launch_bounds__(256)
void ln_bf16(const float* __restrict__ x, const float* __restrict__ g,
             const float* __restrict__ bb, unsigned short* __restrict__ out) {
    __shared__ float red[2][8];
    int row = blockIdx.x, t = threadIdx.x;
    const float* xr = x + (size_t)row * DIMF;
    float a = xr[t], c = xr[t + 256];
    float s = a + c, ss = a * a + c * c;
#pragma unroll
    for (int m = 16; m; m >>= 1) { s += __shfl_xor(s, m); ss += __shfl_xor(ss, m); }
    int wv = t >> 5, ln = t & 31;
    if (ln == 0) { red[0][wv] = s; red[1][wv] = ss; }
    __syncthreads();
    float st = 0.f, sst = 0.f;
#pragma unroll
    for (int k = 0; k < 8; ++k) { st += red[0][k]; sst += red[1][k]; }
    float mu = st * (1.f / DIMF);
    float var = sst * (1.f / DIMF) - mu * mu;
    float rs = rsqrtf(var + EPSV);
    out[(size_t)row * DIMF + t]       = f2bf((a - mu) * rs * g[t] + bb[t]);
    out[(size_t)row * DIMF + t + 256] = f2bf((c - mu) * rs * g[t + 256] + bb[t + 256]);
}

// ---------------------------------------------------------------- bf16 GEMM (WMMA)
// C[M=4096,512] = A[M,512](bf16) x W[512,512]  (W given TRANSPOSED: WbT[n][k])
// mode 0: store bf16 to [B,H,SEQ,64]   (head-major)
// mode 1: store bf16 to [B,H,64,SEQ]   (head-transposed)
// mode 2: store f32 (+bias) to [M,512]
__global__ __launch_bounds__(256)
void gemm_bf16(const unsigned short* __restrict__ A,
               const unsigned short* __restrict__ WbT,
               const float* __restrict__ bias,
               void* __restrict__ dst, int mode) {
    __shared__ unsigned short As[128 * 40];
    __shared__ unsigned short Bs[64 * 40];
    int t = threadIdx.x, lane = t & 31, wv = t >> 5;
    int m0 = blockIdx.x * 128, n0 = blockIdx.y * 64;
    int wm = wv >> 1, wn = wv & 1;
    int lm = lane & 15, hi = lane >> 4;
    v8f c[2][2] = {{{}, {}}, {{}, {}}};

    for (int k0 = 0; k0 < DIMF; k0 += 32) {
#pragma unroll
        for (int it = 0; it < 2; ++it) {               // A tile: 128x32 via async b128
            int v = t + 256 * it;
            int r = v >> 2, kc = (v & 3) * 8;
            async_b128(&A[(size_t)(m0 + r) * DIMF + k0 + kc], &As[r * 40 + kc]);
        }
        {                                              // B tile: 64x32 (pre-transposed)
            int nn = t >> 2, kc = (t & 3) * 8;
            async_b128(&WbT[(size_t)(n0 + nn) * DIMF + k0 + kc], &Bs[nn * 40 + kc]);
        }
        async_wait0();
        __syncthreads();
        Frag af[2], bf[2];
#pragma unroll
        for (int tm = 0; tm < 2; ++tm) {
            int rb = (32 * wm + 16 * tm + lm) * 40 + hi * 8;
            af[tm].q[0] = *(const uint4*)&As[rb];
            af[tm].q[1] = *(const uint4*)&As[rb + 16];
        }
#pragma unroll
        for (int tn = 0; tn < 2; ++tn) {
            int rb = (32 * wn + 16 * tn + lm) * 40 + hi * 8;
            bf[tn].q[0] = *(const uint4*)&Bs[rb];
            bf[tn].q[1] = *(const uint4*)&Bs[rb + 16];
        }
#pragma unroll
        for (int tm = 0; tm < 2; ++tm)
#pragma unroll
            for (int tn = 0; tn < 2; ++tn)
                c[tm][tn] = wmma_bf16(af[tm], bf[tn], c[tm][tn]);
        __syncthreads();
    }
#pragma unroll
    for (int tm = 0; tm < 2; ++tm)
#pragma unroll
        for (int tn = 0; tn < 2; ++tn)
#pragma unroll
            for (int r = 0; r < 8; ++r) {
                int gm = m0 + 32 * wm + 16 * tm + hi * 8 + r;
                int gc = n0 + 32 * wn + 16 * tn + lm;
                float vv = c[tm][tn][r];
                if (mode == 2) {
                    ((float*)dst)[(size_t)gm * DIMF + gc] = vv + bias[gc];
                } else {
                    int b = gm >> 11, i = gm & (SEQ - 1);
                    int h = gc >> 6,  d = gc & 63;
                    unsigned short hv = f2bf(vv);
                    if (mode == 0)
                        ((unsigned short*)dst)[(((size_t)b * HEADS + h) * SEQ + i) * DHEAD + d] = hv;
                    else
                        ((unsigned short*)dst)[(((size_t)b * HEADS + h) * DHEAD + d) * SEQ + i] = hv;
                }
            }
}

// ---------------------------------------------------------------- softmax stats
// Row-softmax stats of sim = (Aq . Bq^T) * SCL over the Bq index.
// Called (qk,cqk) for row softmax and (cqk,qk) for column softmax.
__global__ __launch_bounds__(256)
void attn_stats(const unsigned short* __restrict__ Aq,   // [B,H,SEQ,64]
                const unsigned short* __restrict__ Bq,   // [B,H,SEQ,64]
                float* __restrict__ omax, float* __restrict__ osum) {
    int t = threadIdx.x, lane = t & 31, wv = t >> 5;
    int bh = blockIdx.y;
    int i0 = blockIdx.x * 128 + wv * 16;
    int lm = lane & 15, hi = lane >> 4;
    const unsigned short* Ap = Aq + (size_t)bh * SEQ * DHEAD;
    const unsigned short* Bp = Bq + (size_t)bh * SEQ * DHEAD;

    Frag aq[2];
#pragma unroll
    for (int hh = 0; hh < 2; ++hh) {
        const unsigned short* p = Ap + (size_t)(i0 + lm) * DHEAD + 32 * hh + hi * 8;
        aq[hh].q[0] = *(const uint4*)p;
        aq[hh].q[1] = *(const uint4*)(p + 16);
    }
    float rmax[8], rsum[8];
#pragma unroll
    for (int r = 0; r < 8; ++r) { rmax[r] = -3.0e38f; rsum[r] = 0.f; }

    for (int j0 = 0; j0 < SEQ; j0 += 16) {
        if (j0 + 16 < SEQ)
            __builtin_prefetch(Bp + (size_t)(j0 + 16 + lm) * DHEAD, 0, 1);
        Frag bq[2];
#pragma unroll
        for (int hh = 0; hh < 2; ++hh) {
            const unsigned short* p = Bp + (size_t)(j0 + lm) * DHEAD + 32 * hh + hi * 8;
            bq[hh].q[0] = *(const uint4*)p;
            bq[hh].q[1] = *(const uint4*)(p + 16);
        }
        v8f c = {};
        c = wmma_bf16(aq[0], bq[0], c);
        c = wmma_bf16(aq[1], bq[1], c);
#pragma unroll
        for (int r = 0; r < 8; ++r) {
            float s = c[r] * SCL;
            float tm = s;
            tm = fmaxf(tm, __shfl_xor(tm, 1));
            tm = fmaxf(tm, __shfl_xor(tm, 2));
            tm = fmaxf(tm, __shfl_xor(tm, 4));
            tm = fmaxf(tm, __shfl_xor(tm, 8));
            float nm = fmaxf(rmax[r], tm);
            float e = __expf(s - nm);
            e += __shfl_xor(e, 1);
            e += __shfl_xor(e, 2);
            e += __shfl_xor(e, 4);
            e += __shfl_xor(e, 8);
            rsum[r] = rsum[r] * __expf(rmax[r] - nm) + e;
            rmax[r] = nm;
        }
    }
    if (lm == 0) {
#pragma unroll
        for (int r = 0; r < 8; ++r) {
            int idx = bh * SEQ + i0 + hi * 8 + r;
            omax[idx] = rmax[r];
            osum[idx] = rsum[r];
        }
    }
}

// ---------------------------------------------------------------- P @ V (WMMA)
// out[i,:] = (1/rsum[i]) * sum_j exp(sim[i,j]*SCL - rmax[i]) * Vt[:,j]
// Vt is [B,H,64,SEQ]; output merged-head bf16 [B,SEQ,512].
__global__ __launch_bounds__(256)
void attn_av(const unsigned short* __restrict__ Aq,
             const unsigned short* __restrict__ Bq,
             const unsigned short* __restrict__ Vt,
             const float* __restrict__ rmaxg, const float* __restrict__ rsumg,
             unsigned short* __restrict__ outm) {
    __shared__ unsigned short Pl[8][16 * 40];
    int t = threadIdx.x, lane = t & 31, wv = t >> 5;
    int bh = blockIdx.y;
    int b = bh >> 3, h = bh & 7;
    int i0 = blockIdx.x * 128 + wv * 16;
    int lm = lane & 15, hi = lane >> 4;
    const unsigned short* Ap = Aq + (size_t)bh * SEQ * DHEAD;
    const unsigned short* Bp = Bq + (size_t)bh * SEQ * DHEAD;
    const unsigned short* Vp = Vt + (size_t)bh * DHEAD * SEQ;

    Frag aq[2];
#pragma unroll
    for (int hh = 0; hh < 2; ++hh) {
        const unsigned short* p = Ap + (size_t)(i0 + lm) * DHEAD + 32 * hh + hi * 8;
        aq[hh].q[0] = *(const uint4*)p;
        aq[hh].q[1] = *(const uint4*)(p + 16);
    }
    float rm[8], ir[8];
#pragma unroll
    for (int r = 0; r < 8; ++r) {
        int idx = bh * SEQ + i0 + hi * 8 + r;
        rm[r] = rmaxg[idx];
        ir[r] = 1.f / rsumg[idx];
    }
    v8f acc[4] = {{}, {}, {}, {}};

    for (int j0 = 0; j0 < SEQ; j0 += 32) {
        if (j0 + 32 < SEQ) {
            __builtin_prefetch(Bp + (size_t)(j0 + 32 + lm) * DHEAD, 0, 1);
            __builtin_prefetch(Vp + (size_t)lm * SEQ + j0 + 32, 0, 1);
        }
#pragma unroll
        for (int half = 0; half < 2; ++half) {
            int jc = j0 + 16 * half;
            Frag bq[2];
#pragma unroll
            for (int hh = 0; hh < 2; ++hh) {
                const unsigned short* p = Bp + (size_t)(jc + lm) * DHEAD + 32 * hh + hi * 8;
                bq[hh].q[0] = *(const uint4*)p;
                bq[hh].q[1] = *(const uint4*)(p + 16);
            }
            v8f c = {};
            c = wmma_bf16(aq[0], bq[0], c);
            c = wmma_bf16(aq[1], bq[1], c);
#pragma unroll
            for (int r = 0; r < 8; ++r) {
                float e = __expf(c[r] * SCL - rm[r]);
                Pl[wv][(hi * 8 + r) * 40 + 16 * half + lm] = f2bf(e);
            }
        }
        __syncthreads();
        Frag pa;
        int rb = lm * 40 + hi * 8;
        pa.q[0] = *(const uint4*)&Pl[wv][rb];
        pa.q[1] = *(const uint4*)&Pl[wv][rb + 16];
#pragma unroll
        for (int dt = 0; dt < 4; ++dt) {
            Frag bv;
            const unsigned short* p = Vp + (size_t)(dt * 16 + lm) * SEQ + j0 + hi * 8;
            bv.q[0] = *(const uint4*)p;
            bv.q[1] = *(const uint4*)(p + 16);
            acc[dt] = wmma_bf16(pa, bv, acc[dt]);
        }
        __syncthreads();
    }
#pragma unroll
    for (int dt = 0; dt < 4; ++dt)
#pragma unroll
        for (int r = 0; r < 8; ++r) {
            int i = i0 + hi * 8 + r;
            int col = h * DHEAD + dt * 16 + lm;
            outm[((size_t)b * SEQ + i) * DIMF + col] = f2bf(acc[dt][r] * ir[r]);
        }
}

// ---------------------------------------------------------------- launcher
extern "C" void kernel_launch(void* const* d_in, const int* in_sizes, int n_in,
                              void* d_out, int out_size, void* d_ws, size_t ws_size,
                              hipStream_t stream) {
    (void)in_sizes; (void)n_in; (void)out_size; (void)ws_size;
    const float* x      = (const float*)d_in[0];
    const float* ctx    = (const float*)d_in[1];
    const float* g_x    = (const float*)d_in[2];
    const float* b_x    = (const float*)d_in[3];
    const float* g_c    = (const float*)d_in[4];
    const float* b_c    = (const float*)d_in[5];
    const float* W_qk   = (const float*)d_in[6];
    const float* W_cqk  = (const float*)d_in[7];
    const float* W_v    = (const float*)d_in[8];
    const float* W_cv   = (const float*)d_in[9];
    const float* W_out  = (const float*)d_in[10];
    const float* b_out  = (const float*)d_in[11];
    const float* W_cout = (const float*)d_in[12];
    const float* b_cout = (const float*)d_in[13];

    char* ws = (char*)d_ws;
    size_t off = 0;
    auto alloc = [&](size_t bytes) -> void* {
        void* p = ws + off;
        off += (bytes + 255) & ~(size_t)255;
        return p;
    };
    const size_t act = (size_t)ROWS * DIMF * 2;           // 4 MB bf16 activations
    unsigned short* xn   = (unsigned short*)alloc(act);
    unsigned short* cn   = (unsigned short*)alloc(act);
    unsigned short* qk   = (unsigned short*)alloc(act);   // [B,H,SEQ,64]
    unsigned short* cqk  = (unsigned short*)alloc(act);   // [B,H,SEQ,64]
    unsigned short* vT   = (unsigned short*)alloc(act);   // [B,H,64,SEQ]
    unsigned short* cvT  = (unsigned short*)alloc(act);   // [B,H,64,SEQ]
    const size_t wb = (size_t)DIMF * DIMF * 2;
    unsigned short* Wqkb   = (unsigned short*)alloc(wb);  // all transposed [n][k]
    unsigned short* Wcqkb  = (unsigned short*)alloc(wb);
    unsigned short* Wvb    = (unsigned short*)alloc(wb);
    unsigned short* Wcvb   = (unsigned short*)alloc(wb);
    unsigned short* Woutb  = (unsigned short*)alloc(wb);
    unsigned short* Wcoutb = (unsigned short*)alloc(wb);
    const size_t st = (size_t)BATCH * HEADS * SEQ * 4;
    float* rmax = (float*)alloc(st);
    float* rsum = (float*)alloc(st);
    float* cmax = (float*)alloc(st);
    float* csum = (float*)alloc(st);
    unsigned short* outm = (unsigned short*)alloc(act);
    unsigned short* ctxm = (unsigned short*)alloc(act);

    const int nW = DIMF * DIMF;
    dim3 cb(nW / 256);
    cvt_w_transpose<<<cb, 256, 0, stream>>>(W_qk,   Wqkb);
    cvt_w_transpose<<<cb, 256, 0, stream>>>(W_cqk,  Wcqkb);
    cvt_w_transpose<<<cb, 256, 0, stream>>>(W_v,    Wvb);
    cvt_w_transpose<<<cb, 256, 0, stream>>>(W_cv,   Wcvb);
    cvt_w_transpose<<<cb, 256, 0, stream>>>(W_out,  Woutb);
    cvt_w_transpose<<<cb, 256, 0, stream>>>(W_cout, Wcoutb);

    ln_bf16<<<ROWS, 256, 0, stream>>>(x,   g_x, b_x, xn);
    ln_bf16<<<ROWS, 256, 0, stream>>>(ctx, g_c, b_c, cn);

    dim3 gg(ROWS / 128, DIMF / 64);
    gemm_bf16<<<gg, 256, 0, stream>>>(xn, Wqkb,  nullptr, (void*)qk,  0);
    gemm_bf16<<<gg, 256, 0, stream>>>(cn, Wcqkb, nullptr, (void*)cqk, 0);
    gemm_bf16<<<gg, 256, 0, stream>>>(xn, Wvb,   nullptr, (void*)vT,  1);
    gemm_bf16<<<gg, 256, 0, stream>>>(cn, Wcvb,  nullptr, (void*)cvT, 1);

    dim3 ga(SEQ / 128, BATCH * HEADS);
    attn_stats<<<ga, 256, 0, stream>>>(qk,  cqk, rmax, rsum);   // softmax over j
    attn_stats<<<ga, 256, 0, stream>>>(cqk, qk,  cmax, csum);   // softmax over i

    attn_av<<<ga, 256, 0, stream>>>(qk,  cqk, cvT, rmax, rsum, outm);
    attn_av<<<ga, 256, 0, stream>>>(cqk, qk,  vT,  cmax, csum, ctxm);

    float* out0 = (float*)d_out;
    float* out1 = out0 + (size_t)ROWS * DIMF;
    gemm_bf16<<<gg, 256, 0, stream>>>(outm, Woutb,  b_out,  (void*)out0, 2);
    gemm_bf16<<<gg, 256, 0, stream>>>(ctxm, Wcoutb, b_cout, (void*)out1, 2);
}
